// preDeepMD_43198781063816
// MI455X (gfx1250) — matrix operand order repaired and checked by reading, not compiled, not run on hardware
//
#include <hip/hip_runtime.h>
#include <hip/hip_bf16.h>
#include <math.h>

typedef _Float16 f16;
typedef __attribute__((ext_vector_type(16))) _Float16 v16h;
typedef __attribute__((ext_vector_type(8)))  float    v8f;

// Branch-free tanh on raw CDNA5 transcendentals:
// tanh(x) = 1 - 2/(e^{2x}+1), e^{2x} = 2^{x * 2*log2(e)}.
// fast_tanh(0) == 0 exactly; saturates exactly at +-1; no EXEC divergence.
__device__ __forceinline__ float fast_tanh(float x) {
  float e = __builtin_amdgcn_exp2f(x * 2.8853900817779268f);
  return 1.0f - 2.0f * __builtin_amdgcn_rcpf(e + 1.0f);
}

// ---------------------------------------------------------------------------
// WMMA 16x16x32 f16 fragment loaders (wave32), CDNA5 ISA 7.12.2 layouts,
// fully vectorized as b128 loads.
// A (16x32 row-major, ld multiple of 8): lane r=lane&15, hi=lane>>4 holds
//   K = hi*8 + {0..7} and hi*8 + {16..23}  -> two contiguous 16B runs.
// B is consumed from a TRANSPOSED weight array Wt[N][K] (ldk multiple of 8):
//   lane col c=lane&15 holds K = hi*8 + {0..7} and hi*8 + {16..23} -> two 16B runs.
// C/D (16x16 f32): VGPR j -> M = j + hi*8, N = lane&15.
// ---------------------------------------------------------------------------
__device__ __forceinline__ v16h load_a_frag(const f16* base, int ld, int lane) {
  int r = lane & 15, hi = lane >> 4;
  const f16* p = base + (size_t)r * ld + hi * 8;
  union { uint4 u[2]; v16h h; } u;
  u.u[0] = *(const uint4*)(p);
  u.u[1] = *(const uint4*)(p + 16);
  return u.h;
}

__device__ __forceinline__ v16h load_bT_frag(const f16* Wt, int ldk, int col0,
                                             int k0, int lane) {
  int c = lane & 15, hi = lane >> 4;
  const f16* p = Wt + (size_t)(col0 + c) * ldk + k0 + hi * 8;
  union { uint4 u[2]; v16h h; } u;
  u.u[0] = *(const uint4*)(p);
  u.u[1] = *(const uint4*)(p + 16);
  return u.h;
}

// Async global -> LDS 16-byte copy (CDNA5 GLOBAL_LOAD_ASYNC_TO_LDS_B128,
// tracked by ASYNCcnt). lds/global addresses per-lane.
__device__ __forceinline__ void async_g2l_b128(uint32_t lds_addr, uint64_t gaddr) {
  asm volatile("global_load_async_to_lds_b128 %0, %1, off"
               :: "v"(lds_addr), "v"(gaddr) : "memory");
}
__device__ __forceinline__ void wait_asynccnt0() {
  asm volatile("s_wait_asynccnt 0" ::: "memory");
}

// ---------------------------------------------------------------------------
// Weight convert: f32 [K x N] -> f16 TRANSPOSED + zero-padded [dstN x dstK].
// Padded entries are exact zeros => padded K/N dims contribute nothing.
// ---------------------------------------------------------------------------
__global__ void convert_pad_t(const float* __restrict__ src, f16* __restrict__ dst,
                              int srcK, int srcN, int dstN, int dstK) {
  int idx = blockIdx.x * blockDim.x + threadIdx.x;
  int total = dstN * dstK;
  if (idx >= total) return;
  int n = idx / dstK, k = idx % dstK;
  float v = (k < srcK && n < srcN) ? src[k * srcN + n] : 0.0f;
  dst[idx] = (f16)v;
}

// Bias pad: f32 [n] -> f32 [npad], zeros beyond n.
__global__ void pad_bias(const float* __restrict__ src, float* __restrict__ dst,
                         int n, int npad) {
  int idx = blockIdx.x * blockDim.x + threadIdx.x;
  if (idx >= npad) return;
  dst[idx] = (idx < n) ? src[idx] : 0.0f;
}

// ---------------------------------------------------------------------------
// Kernel 1: one workgroup per (b,n). 128 threads = 4 waves.
// ---------------------------------------------------------------------------
__global__ __launch_bounds__(128)
void descriptor_kernel(const float* __restrict__ imageDR,
                       const float* __restrict__ ew0, const float* __restrict__ eb0,
                       const f16*  __restrict__ W1t, const float* __restrict__ eb1p,
                       const f16*  __restrict__ W2t, const float* __restrict__ eb2p,
                       f16* __restrict__ DRout) {
  __shared__ __align__(16) float Ri[128 * 4];   // [k][4] : S, coef*xyz
  __shared__ __align__(16) f16   H1[128 * 64];  // layer1 activations (padded)
  // Phase-overlapped region: H0 (phase A) aliases the front of G (phase B).
  __shared__ __align__(16) char Xraw[128 * 112 * 2 + 64 * 4 + 400 * 4];
  f16*   H0 = (f16*)Xraw;                       // 128x32  (phase A)
  f16*   G  = (f16*)Xraw;                       // 128x112 (phase B)
  float* tA = (float*)(Xraw + 128 * 112 * 2);   // 16x4
  float* tB = tA + 64;                          // 4x100

  const int t    = threadIdx.x;
  const int bn   = blockIdx.x;
  const int lane = t & 31;
  const int w    = t >> 5;

  // ---- per-neighbor cutoff + layer0 (1 -> 25, tanh) ----
  {
    const float4 p4 = *(const float4*)(imageDR + ((size_t)bn * 128 + t) * 4);
    float x = p4.x, y = p4.y, z = p4.z, fl = p4.w;
    float rij  = x * x + y * y + z * z;
    bool  mask = fl > 0.0f;
    bool  lt10 = rij < 10.0f;
    float safe = (rij == 0.0f) ? 1.0f : rij;
    // 0.5*cos(pi*(r-10)/15)+0.5 via hardware v_cos_f32 (arg in turns):
    float cosv = 0.5f * __builtin_amdgcn_cosf((rij - 10.0f) * (1.0f / 30.0f)) + 0.5f;
    float S;
    if (mask && lt10)              S = __builtin_amdgcn_rcpf(safe);
    else if (!lt10 && rij < 25.0f) S = cosv;
    else                           S = 0.0f;
    float coef = mask ? S * __builtin_amdgcn_rcpf(safe) : 0.0f;
    Ri[t * 4 + 0] = S;
    Ri[t * 4 + 1] = coef * x;
    Ri[t * 4 + 2] = coef * y;
    Ri[t * 4 + 3] = coef * z;

    union { uint4 u[4]; f16 h[32]; } row;
#pragma unroll 1
    for (int j = 0; j < 25; ++j) row.h[j] = (f16)fast_tanh(S * ew0[j] + eb0[j]);
#pragma unroll
    for (int j = 25; j < 32; ++j) row.h[j] = (f16)0;
    uint4* dst = (uint4*)(H0 + t * 32);
#pragma unroll
    for (int q = 0; q < 4; ++q) dst[q] = row.u[q];
  }
  __syncthreads();

  // ---- layer1: (128x32) @ (32x64) via WMMA; 8 row-tiles x 4 col-tiles ----
  // Padded cols: B cols are zero => c==0, bias pad is zero => tanh(0)==0, so
  // the epilogue is fully unconditional (no EXEC divergence).
#pragma unroll 1
  for (int i = 0; i < 8; ++i) {
    int tid = w * 8 + i;
    int mt = tid >> 2, nt = tid & 3;
    v16h a = load_a_frag(H0 + mt * 16 * 32, 32, lane);
    v16h b = load_bT_frag(W1t, 32, nt * 16, 0, lane);
    v8f  c = {};
    c = __builtin_amdgcn_wmma_f32_16x16x32_f16(false, a, false, b, (short)0, c, false, false);
    int cc = lane & 15, hi = lane >> 4;
    int col = nt * 16 + cc;
    float bias = eb1p[col];
#pragma unroll
    for (int j = 0; j < 8; ++j) {
      int row = mt * 16 + hi * 8 + j;
      H1[row * 64 + col] = (f16)fast_tanh(c[j] + bias);
    }
  }
  __syncthreads();

  // ---- layer2: (128x64) @ (64x112) via WMMA; 8 x 7 tiles, 2 K-steps ----
#pragma unroll 1
  for (int i = 0; i < 14; ++i) {
    int tid = w * 14 + i;
    int mt = tid / 7, nt = tid % 7;
    v8f c = {};
#pragma unroll
    for (int s = 0; s < 2; ++s) {
      int k0 = s * 32;
      v16h a = load_a_frag(H1 + mt * 16 * 64 + k0, 64, lane);
      v16h b = load_bT_frag(W2t, 64, nt * 16, k0, lane);
      c = __builtin_amdgcn_wmma_f32_16x16x32_f16(false, a, false, b, (short)0, c, false, false);
    }
    int cc = lane & 15, hi = lane >> 4;
    int col = nt * 16 + cc;
    float bias = eb2p[col];
#pragma unroll
    for (int j = 0; j < 8; ++j) {
      int row = mt * 16 + hi * 8 + j;
      G[row * 112 + col] = (f16)fast_tanh(c[j] + bias);
    }
  }
  __syncthreads();

  // ---- tmpA (16x4) = G16^T @ Ri ; tmpB (4x100) = Ri^T @ G ----
  if (t < 64) {
    int m = t >> 2, f = t & 3;
    float acc = 0.0f;
    for (int j = 0; j < 128; ++j) acc += (float)G[j * 112 + m] * Ri[j * 4 + f];
    tA[t] = acc;
  }
  for (int idx = t; idx < 400; idx += 128) {
    int f = idx / 100, m = idx % 100;
    float acc = 0.0f;
    for (int j = 0; j < 128; ++j) acc += Ri[j * 4 + f] * (float)G[j * 112 + m];
    tB[idx] = acc;
  }
  __syncthreads();

  // ---- DR = tmpA @ tmpB -> 16x100 row-major, stored f16 ----
  f16* drp = DRout + (size_t)bn * 1600;
  for (int idx = t; idx < 1600; idx += 128) {
    int m = idx / 100, h = idx % 100;
    float acc = 0.0f;
#pragma unroll
    for (int f = 0; f < 4; ++f) acc += tA[m * 4 + f] * tB[f * 100 + h];
    drp[idx] = (f16)acc;
  }
}

// ---------------------------------------------------------------------------
// Kernel 2: fitting GEMM, Out = tanh(A @ W + bias).
// 4 waves / block; each wave owns one 16x16 tile (4 stacked row-tiles share
// one 16-col weight tile). Per K-step the 16x32 W^T tile is staged to LDS by
// wave 0 via GLOBAL_LOAD_ASYNC_TO_LDS_B128, double-buffered (ASYNCcnt).
// ---------------------------------------------------------------------------
__global__ __launch_bounds__(128)
void fit_gemm_tanh(const f16* __restrict__ A, int lda,
                   const f16* __restrict__ Wt, int ldk,
                   const float* __restrict__ bias,
                   f16* __restrict__ Out, int ldo, int ksteps) {
  __shared__ __align__(16) f16 Wbuf[2][16][32];   // [buf][n-in-tile][k] : 2KB

  const int t    = threadIdx.x;
  const int lane = t & 31;
  const int w    = t >> 5;
  const int rt   = blockIdx.x * 4 + w;            // row tile (16 rows)
  const int ct   = blockIdx.y;                    // col tile (16 cols)
  const int col0 = ct * 16;

  const f16* Ab = A + (size_t)rt * 16 * lda;

  // per-lane staging coords for the W tile
  const int srow = lane & 15, shi = lane >> 4;
  const uint64_t gbase = (uint64_t)(uintptr_t)(Wt + (size_t)(col0 + srow) * ldk + shi * 16);

  // prologue: stage k-step 0 into buffer 0
  if (w == 0) {
    uint32_t l0 = (uint32_t)(uintptr_t)&Wbuf[0][srow][shi * 16];
    async_g2l_b128(l0, gbase);
    async_g2l_b128(l0 + 16, gbase + 16);
    wait_asynccnt0();
  }
  __syncthreads();

  v8f c = {};
#pragma unroll 1
  for (int s = 0; s < ksteps; ++s) {
    const int cur = s & 1;
    const int k0  = s * 32;
    if (w == 0 && s + 1 < ksteps) {
      uint32_t ln = (uint32_t)(uintptr_t)&Wbuf[cur ^ 1][srow][shi * 16];
      uint64_t gn = gbase + (uint64_t)(k0 + 32) * sizeof(f16);
      async_g2l_b128(ln, gn);
      async_g2l_b128(ln + 16, gn + 16);
    }
    __builtin_prefetch(Ab + k0 + 64, 0, 1);       // global_prefetch_b8 on A stream
    v16h a = load_a_frag(Ab + k0, lda, lane);
    v16h b = load_bT_frag(&Wbuf[cur][0][0], 32, 0, 0, lane);
    c = __builtin_amdgcn_wmma_f32_16x16x32_f16(false, a, false, b, (short)0, c, false, false);
    if (w == 0 && s + 1 < ksteps) wait_asynccnt0();
    __syncthreads();
  }

  int cc = lane & 15, hi = lane >> 4;
  int col = col0 + cc;
  float bv = bias[col];
#pragma unroll
  for (int j = 0; j < 8; ++j) {
    int row = rt * 16 + hi * 8 + j;
    Out[(size_t)row * ldo + col] = (f16)fast_tanh(c[j] + bv);
  }
}

// ---------------------------------------------------------------------------
// Kernel 3: Ei[row] = dot(act3[row, :240], fw3) + fb3  (one wave per row)
// ---------------------------------------------------------------------------
__global__ __launch_bounds__(32)
void ei_kernel(const f16* __restrict__ act3, const float* __restrict__ fw3,
               const float* __restrict__ fb3, float* __restrict__ out) {
  int row = blockIdx.x;
  int lane = threadIdx.x;
  float acc = 0.0f;
  for (int j = lane; j < 240; j += 32)
    acc += (float)act3[(size_t)row * 256 + j] * fw3[j];
#pragma unroll
  for (int o = 16; o > 0; o >>= 1) acc += __shfl_down(acc, o, 32);
  if (lane == 0) out[8 + row] = acc + fb3[0];
}

// ---------------------------------------------------------------------------
// Kernel 4: Etot[b] = sum_n Ei[b,n]  (deterministic LDS tree reduce)
// ---------------------------------------------------------------------------
__global__ __launch_bounds__(256)
void etot_kernel(float* __restrict__ out) {
  __shared__ float red[256];
  int b = blockIdx.x, t = threadIdx.x;
  float acc = 0.0f;
  for (int i = t; i < 512; i += 256) acc += out[8 + b * 512 + i];
  red[t] = acc;
  __syncthreads();
  for (int s = 128; s > 0; s >>= 1) {
    if (t < s) red[t] += red[t + s];
    __syncthreads();
  }
  if (t == 0) out[b] = red[0];
}

// ---------------------------------------------------------------------------
// Host-side launch. Workspace (f16 elements), weights stored TRANSPOSED:
//   W1t @0 (64x32) | W2t @2048 (112x64) | F0t @9216 (240x1600)
//   F1t @393216 (240x256) | F2t @454656 (240x256)
//   DR @516096 (4096x1600) | A1 @7069696 | A2 @8118272 | A3 @9166848 (4096x256)
//   then f32 padded biases: eb1p (64), eb2p (112) at half-offset 10,215,424.
//   total ~ 20.43 MB
// ---------------------------------------------------------------------------
extern "C" void kernel_launch(void* const* d_in, const int* in_sizes, int n_in,
                              void* d_out, int out_size, void* d_ws, size_t ws_size,
                              hipStream_t stream) {
  (void)in_sizes; (void)n_in; (void)out_size; (void)ws_size;
  const float* imageDR = (const float*)d_in[0];
  const float* ew0 = (const float*)d_in[1];
  const float* eb0 = (const float*)d_in[2];
  const float* ew1 = (const float*)d_in[3];
  const float* eb1 = (const float*)d_in[4];
  const float* ew2 = (const float*)d_in[5];
  const float* eb2 = (const float*)d_in[6];
  const float* fw0 = (const float*)d_in[7];
  const float* fb0 = (const float*)d_in[8];
  const float* fw1 = (const float*)d_in[9];
  const float* fb1 = (const float*)d_in[10];
  const float* fw2 = (const float*)d_in[11];
  const float* fb2 = (const float*)d_in[12];
  const float* fw3 = (const float*)d_in[13];
  const float* fb3 = (const float*)d_in[14];
  float* out = (float*)d_out;

  f16* H   = (f16*)d_ws;
  f16* W1t = H;
  f16* W2t = H + 2048;
  f16* F0t = H + 9216;
  f16* F1t = H + 393216;
  f16* F2t = H + 454656;
  f16* DR  = H + 516096;
  f16* A1  = H + 7069696;
  f16* A2  = H + 8118272;
  f16* A3  = H + 9166848;
  float* eb1p = (float*)(H + 10215424);   // 64 f32
  float* eb2p = eb1p + 64;                // 112 f32

  auto cvt = [&](const float* s, f16* d, int sk, int sn, int dn, int dk) {
    int total = dn * dk;
    convert_pad_t<<<(total + 255) / 256, 256, 0, stream>>>(s, d, sk, sn, dn, dk);
  };
  cvt(ew1, W1t, 25, 50, 64, 32);
  cvt(ew2, W2t, 50, 100, 112, 64);
  cvt(fw0, F0t, 1600, 240, 240, 1600);
  cvt(fw1, F1t, 240, 240, 240, 256);
  cvt(fw2, F2t, 240, 240, 240, 256);
  pad_bias<<<1, 64, 0, stream>>>(eb1, eb1p, 50, 64);
  pad_bias<<<1, 128, 0, stream>>>(eb2, eb2p, 100, 112);

  descriptor_kernel<<<4096, 128, 0, stream>>>(imageDR, ew0, eb0, W1t, eb1p, W2t, eb2p, DR);

  dim3 g1(64, 15);
  fit_gemm_tanh<<<g1, 128, 0, stream>>>(DR, 1600, F0t, 1600, fb0, A1, 256, 50);
  fit_gemm_tanh<<<g1, 128, 0, stream>>>(A1, 256, F1t, 256, fb1, A2, 256, 8);
  fit_gemm_tanh<<<g1, 128, 0, stream>>>(A2, 256, F2t, 256, fb2, A3, 256, 8);

  ei_kernel<<<4096, 32, 0, stream>>>(A3, fw3, fb3, out);
  etot_kernel<<<8, 256, 0, stream>>>(out);
}
